// gCNN_37598143709502
// MI455X (gfx1250) — matrix operand order repaired and checked by reading, not compiled, not run on hardware
//
#include <hip/hip_runtime.h>
#include <hip/hip_bf16.h>

// ---------------------------------------------------------------------------
// gCNN on icosphere mesh: 4x (gather-interp conv -> GEMM) with BN+ReLU between.
// GEMMs run on v_wmma_f32_16x16x32_f16 (wave32 CDNA5 WMMA). The [16,K] agg
// tile is built in LDS as f16 with ping-pong double buffering so the next
// chunk's gather (random L2 reads) overlaps the current chunk's WMMAs.
// Weights are pre-converted to zero-padded f16 in workspace.
// ---------------------------------------------------------------------------

typedef __attribute__((ext_vector_type(16))) _Float16 v16h;
typedef __attribute__((ext_vector_type(8)))  _Float16 v8h;
typedef __attribute__((ext_vector_type(8)))  float    v8f;

#define N_NODES 10242
#define BN_EPS  1e-5f

// ---------------------------------------------------------------------------
// Weight convert + zero-pad: W [rows, cols] f32 -> Wh [rowsP, colsP] f16
// ---------------------------------------------------------------------------
__global__ void cvt_pad_kernel(const float* __restrict__ W, _Float16* __restrict__ Wh,
                               int rows, int cols, int rowsP, int colsP)
{
    int i = blockIdx.x * blockDim.x + threadIdx.x;
    if (i < rowsP * colsP) {
        int r = i / colsP, c = i % colsP;
        float v = (r < rows && c < cols) ? W[r * cols + c] : 0.0f;
        Wh[i] = (_Float16)v;
    }
}

// ---------------------------------------------------------------------------
// Conv layer: 16 nodes per block, 8 waves (256 threads), one 16x16 D-tile per
// wave, WMMA 16x16x32 f16 -> f32 accumulate.
//   CIN  : input channels (3, 64, 128)
//   COUT : real output channels
//   COUTP: padded output channels (multiple of 16, NTILE = COUTP/16 <= 8)
// Wh is [COUTP, KSTR] f16 (KSTR = 96 for CIN==3, else 25*CIN), zero padded.
// ---------------------------------------------------------------------------
template<int CIN, int COUT, int COUTP>
__global__ void __launch_bounds__(256)
conv_wmma_kernel(const float* __restrict__ hin,
                 const int*   __restrict__ idx,
                 const float* __restrict__ wgt,
                 const _Float16* __restrict__ Wh,
                 const float* __restrict__ bias,
                 float* __restrict__ out)
{
    constexpr int CHUNK = (CIN == 3) ? 96 : CIN;      // flattened-K per outer iter
    constexpr int KITER = (CIN == 3) ? 1  : 25;       // outer neighborhood loop
    constexpr int KSTR  = (CIN == 3) ? 96 : 25 * CIN; // Wh row stride (halfs)
    constexpr int AS    = CHUNK + 8;                  // LDS row stride (16B-aligned, bank-skewed)
    constexpr int NTILE = COUTP / 16;                 // <= 8

    __shared__ __align__(16) _Float16 agg[2][16 * AS];

    const int tid  = threadIdx.x;
    const int lane = tid & 31;
    const int wv   = tid >> 5;
    const int base = blockIdx.x * 16;

    // ---- gather/interp one K-chunk into LDS buffer `buf` ----
    auto fill = [&](int ko, int buf) {
        if constexpr (CIN == 3) {
            v8h z = {};
            for (int i = tid * 8; i < 16 * AS; i += 256 * 8)
                *(v8h*)(&agg[buf][i]) = z;
            __syncthreads();                           // zero before scatter
            for (int it = tid; it < 16 * 25; it += 256) {
                int m = it / 25, k = it % 25;
                int node = base + m; if (node >= N_NODES) node = N_NODES - 1;
                int off = node * 75 + k * 3;
                int   i0 = idx[off], i1 = idx[off + 1], i2 = idx[off + 2];
                float w0 = wgt[off], w1 = wgt[off + 1], w2 = wgt[off + 2];
                #pragma unroll
                for (int e = 0; e < 3; ++e) {
                    float v = w0 * hin[i0 * 3 + e] + w1 * hin[i1 * 3 + e]
                            + w2 * hin[i2 * 3 + e];
                    agg[buf][m * AS + k * 3 + e] = (_Float16)v;
                }
            }
        } else {
            constexpr int CW = CIN / 16;               // channels per thread (4 or 8)
            int m  = tid >> 4;                         // 16 threads per node row
            int c0 = (tid & 15) * CW;
            int node = base + m; if (node >= N_NODES) node = N_NODES - 1;
            int off = node * 75 + ko * 3;
            int   i0 = idx[off], i1 = idx[off + 1], i2 = idx[off + 2];
            float w0 = wgt[off], w1 = wgt[off + 1], w2 = wgt[off + 2];
            const float* r0 = hin + i0 * CIN;
            const float* r1 = hin + i1 * CIN;
            const float* r2 = hin + i2 * CIN;
            #pragma unroll
            for (int c = c0; c < c0 + CW; ++c) {
                float v = w0 * r0[c] + w1 * r1[c] + w2 * r2[c];
                agg[buf][m * AS + c] = (_Float16)v;
            }
        }
    };

    v8f acc = {};
    const int  mrow = lane & 15;
    const int  asel = (lane < 16) ? 0 : 8;   // A frag: K 0-7/16-23 vs 8-15/24-31
    const int  bsel = (lane < 16) ? 0 : 8;   // B frag: K 0-7/16-23 vs 8-15/24-31
    const bool live = (wv < NTILE);          // wave-uniform: this wave owns a tile

    fill(0, 0);
    __syncthreads();

    for (int ko = 0; ko < KITER; ++ko) {
        const int cur = ko & 1;
        if (ko + 1 < KITER)
            fill(ko + 1, cur ^ 1);           // overlap next gather with WMMAs

        #pragma unroll
        for (int kc = 0; kc < CHUNK; kc += 32) {
            const _Float16* ap = &agg[cur][mrow * AS + kc + asel];
            v8h a0 = *(const v8h*)(ap);
            v8h a1 = *(const v8h*)(ap + 16);
            v16h av = __builtin_shufflevector(a0, a1,
                        0,1,2,3,4,5,6,7,8,9,10,11,12,13,14,15);
            if (live) {
                const int kglob = (CIN == 3) ? kc : (ko * CIN + kc);
                const _Float16* bp = Wh + (wv * 16 + mrow) * KSTR + kglob + bsel;
                v8h b0 = *(const v8h*)(bp);
                v8h b1 = *(const v8h*)(bp + 16);
                v16h bv = __builtin_shufflevector(b0, b1,
                            0,1,2,3,4,5,6,7,8,9,10,11,12,13,14,15);
                acc = __builtin_amdgcn_wmma_f32_16x16x32_f16(
                    false, av, false, bv, (short)0, acc, false, false);
            }
        }
        __syncthreads();                      // fills done + reads done
    }

    // ---- store D tile (+bias) ----
    if (live) {
        const int col = wv * 16 + (lane & 15);
        #pragma unroll
        for (int r = 0; r < 8; ++r) {
            const int m = r + ((lane >= 16) ? 8 : 0);
            const int node = base + m;
            if (node < N_NODES && col < COUT)
                out[node * COUT + col] = acc[r] + bias[col];
        }
    }
}

// ---------------------------------------------------------------------------
// BN stats stage 1: per-block partial sum / sumsq over rows. Deterministic.
// part layout: [nblk][COUT] sums, then [nblk][COUT] sumsqs.
// ---------------------------------------------------------------------------
template<int COUT>
__global__ void __launch_bounds__(256)
bn_reduce1_kernel(const float* __restrict__ x, float* __restrict__ part, int nrows)
{
    constexpr int SUB = 256 / COUT;
    __shared__ float s1[256];
    __shared__ float s2[256];
    const int c   = threadIdx.x % COUT;
    const int sub = threadIdx.x / COUT;
    float a = 0.0f, b = 0.0f;
    for (int r = blockIdx.x * SUB + sub; r < nrows; r += gridDim.x * SUB) {
        float v = x[r * COUT + c];
        a += v; b += v * v;
    }
    s1[threadIdx.x] = a; s2[threadIdx.x] = b;
    __syncthreads();
    if (sub == 0) {
        #pragma unroll
        for (int s = 1; s < SUB; ++s) { a += s1[c + s * COUT]; b += s2[c + s * COUT]; }
        part[blockIdx.x * COUT + c] = a;
        part[(gridDim.x + blockIdx.x) * COUT + c] = b;
    }
}

// BN stats stage 2: finalize mean + rsqrt(var+eps). One block.
template<int COUT>
__global__ void bn_reduce2_kernel(const float* __restrict__ part,
                                  float* __restrict__ stats, int nblk, int nrows)
{
    const int c = threadIdx.x;
    if (c < COUT) {
        float a = 0.0f, b = 0.0f;
        for (int i = 0; i < nblk; ++i) {
            a += part[i * COUT + c];
            b += part[(nblk + i) * COUT + c];
        }
        const float inv_n = 1.0f / (float)nrows;
        const float mean  = a * inv_n;
        const float var   = b * inv_n - mean * mean;
        stats[c]        = mean;
        stats[COUT + c] = rsqrtf(var + BN_EPS);
    }
}

// BN apply + ReLU, in place.
template<int COUT>
__global__ void bn_apply_kernel(float* __restrict__ x, const float* __restrict__ stats,
                                const float* __restrict__ g, const float* __restrict__ be,
                                int nrows)
{
    const int i = blockIdx.x * blockDim.x + threadIdx.x;
    if (i < nrows * COUT) {
        const int c = i % COUT;
        float v = (x[i] - stats[c]) * stats[COUT + c] * g[c] + be[c];
        x[i] = fmaxf(v, 0.0f);
    }
}

// ---------------------------------------------------------------------------
// Host launch
// ---------------------------------------------------------------------------
extern "C" void kernel_launch(void* const* d_in, const int* in_sizes, int n_in,
                              void* d_out, int out_size, void* d_ws, size_t ws_size,
                              hipStream_t stream)
{
    (void)in_sizes; (void)n_in; (void)out_size; (void)ws_size;

    const float* x   = (const float*)d_in[0];
    const int*   idx = (const int*)  d_in[1];
    const float* wgt = (const float*)d_in[2];
    const float* W1  = (const float*)d_in[3];  const float* b1  = (const float*)d_in[4];
    const float* g1  = (const float*)d_in[5];  const float* be1 = (const float*)d_in[6];
    const float* W2  = (const float*)d_in[7];  const float* b2  = (const float*)d_in[8];
    const float* g2  = (const float*)d_in[9];  const float* be2 = (const float*)d_in[10];
    const float* W3  = (const float*)d_in[11]; const float* b3  = (const float*)d_in[12];
    const float* g3  = (const float*)d_in[13]; const float* be3 = (const float*)d_in[14];
    const float* W4  = (const float*)d_in[15]; const float* b4  = (const float*)d_in[16];
    float* out = (float*)d_out;

    // workspace layout
    char* ws = (char*)d_ws;
    size_t off = 0;
    auto take = [&](size_t bytes) -> void* {
        void* p = ws + off;
        off = (off + bytes + 255) & ~(size_t)255;
        return p;
    };
    _Float16* Wh1 = (_Float16*)take((size_t)64  * 96   * 2);
    _Float16* Wh2 = (_Float16*)take((size_t)64  * 1600 * 2);
    _Float16* Wh3 = (_Float16*)take((size_t)128 * 1600 * 2);
    _Float16* Wh4 = (_Float16*)take((size_t)48  * 3200 * 2);
    float* bufA  = (float*)take((size_t)N_NODES * 128 * 4);
    float* bufB  = (float*)take((size_t)N_NODES * 128 * 4);
    float* part  = (float*)take((size_t)64 * 2 * 128 * 4);
    float* stats = (float*)take((size_t)2 * 128 * 4);

    // weight convert + pad
    cvt_pad_kernel<<<(64 * 96   + 255) / 256, 256, 0, stream>>>(W1, Wh1, 64, 75,   64, 96);
    cvt_pad_kernel<<<(64 * 1600 + 255) / 256, 256, 0, stream>>>(W2, Wh2, 64, 1600, 64, 1600);
    cvt_pad_kernel<<<(128* 1600 + 255) / 256, 256, 0, stream>>>(W3, Wh3, 128,1600, 128,1600);
    cvt_pad_kernel<<<(48 * 3200 + 255) / 256, 256, 0, stream>>>(W4, Wh4, 36, 3200, 48, 3200);

    const int NB = (N_NODES + 15) / 16;   // 641 node-tiles
    const int RB = 64;                    // reduction blocks

    // layer 1: x[N,3] -> bufA[N,64], BN+ReLU in place
    conv_wmma_kernel<3, 64, 64><<<NB, 256, 0, stream>>>(x, idx, wgt, Wh1, b1, bufA);
    bn_reduce1_kernel<64><<<RB, 256, 0, stream>>>(bufA, part, N_NODES);
    bn_reduce2_kernel<64><<<1, 128, 0, stream>>>(part, stats, RB, N_NODES);
    bn_apply_kernel<64><<<(N_NODES * 64 + 255) / 256, 256, 0, stream>>>(bufA, stats, g1, be1, N_NODES);

    // layer 2: bufA[N,64] -> bufB[N,64]
    conv_wmma_kernel<64, 64, 64><<<NB, 256, 0, stream>>>(bufA, idx, wgt, Wh2, b2, bufB);
    bn_reduce1_kernel<64><<<RB, 256, 0, stream>>>(bufB, part, N_NODES);
    bn_reduce2_kernel<64><<<1, 128, 0, stream>>>(part, stats, RB, N_NODES);
    bn_apply_kernel<64><<<(N_NODES * 64 + 255) / 256, 256, 0, stream>>>(bufB, stats, g2, be2, N_NODES);

    // layer 3: bufB[N,64] -> bufA[N,128]
    conv_wmma_kernel<64, 128, 128><<<NB, 256, 0, stream>>>(bufB, idx, wgt, Wh3, b3, bufA);
    bn_reduce1_kernel<128><<<RB, 256, 0, stream>>>(bufA, part, N_NODES);
    bn_reduce2_kernel<128><<<1, 128, 0, stream>>>(part, stats, RB, N_NODES);
    bn_apply_kernel<128><<<(N_NODES * 128 + 255) / 256, 256, 0, stream>>>(bufA, stats, g3, be3, N_NODES);

    // layer 4: bufA[N,128] -> out[N,36]
    conv_wmma_kernel<128, 36, 48><<<NB, 256, 0, stream>>>(bufA, idx, wgt, Wh4, b4, out);
}
